// CrossAttention_24309514895437
// MI455X (gfx1250) — compile-verified, matrix operand order checked by reading, not run on hardware
//
#include <hip/hip_runtime.h>
#include <stdint.h>

// ---------------------------------------------------------------------------
// CrossAttention for MI455X (gfx1250): bf16 WMMA everywhere.
// B=4, L=2048, IN=EMBED=1024, H=16, D=64.
// Projections / output proj: 64x64 tiles per wave (16 WMMA / K-step).
// Attention: flash-style, 32 queries per wave, 32 keys per step.
// All matrix kernels are deliberate 1-wave register-fat kernels:
// __launch_bounds__(32, 1) so the allocator can use the full VGPR file.
// ---------------------------------------------------------------------------

#define NB    4
#define NL    2048
#define NIN   1024
#define NH    16
#define ND    64
#define MAXSM 4.6051702f   // log(100)

typedef __attribute__((ext_vector_type(16))) __bf16       bf16x16;
typedef __attribute__((ext_vector_type(8)))  float        f32x8;
typedef __attribute__((ext_vector_type(4)))  float        f32x4;
typedef __attribute__((ext_vector_type(4)))  unsigned int u32x4;

static __device__ __forceinline__ f32x8 wmma_bf16(bf16x16 a, bf16x16 b, f32x8 c) {
    return __builtin_amdgcn_wmma_f32_16x16x32_bf16(
        /*neg_a=*/false, a, /*neg_b=*/false, b,
        /*c_mod=*/(short)0, c, /*reuse_a=*/false, /*reuse_b=*/false);
}

static __device__ __forceinline__ unsigned short bf16_bits(float f) {
    union { float f; uint32_t u; } a; a.f = f;
    return (unsigned short)((a.u + 0x7FFFu + ((a.u >> 16) & 1u)) >> 16);
}
static __device__ __forceinline__ __bf16 to_bf16(float f) {
    union { unsigned short s; __bf16 h; } o; o.s = bf16_bits(f);
    return o.h;
}

// Load one 16x32 bf16 fragment (A-pattern; B uses same pattern indexed by N)
// from a row-major matrix with leading dimension ldm (in elements).
// lane<16 : row = lane,    K chunks {0..7},{16..23}
// lane>=16: row = lane-16, K chunks {8..15},{24..31}
static __device__ __forceinline__ bf16x16 load_frag(const __bf16* base, int ldm) {
    const int lane = threadIdx.x & 31;
    const int r16  = lane & 15;
    const int hi   = lane >> 4;
    const __bf16* p = base + (size_t)r16 * ldm + hi * 8;
    union { bf16x16 v; u32x4 q[2]; } u;
    u.q[0] = *reinterpret_cast<const u32x4*>(p);
    u.q[1] = *reinterpret_cast<const u32x4*>(p + 16);
    return u.v;
}

static __device__ __forceinline__ float rsum16(float v) {
    v += __shfl_xor(v, 1, 32); v += __shfl_xor(v, 2, 32);
    v += __shfl_xor(v, 4, 32); v += __shfl_xor(v, 8, 32);
    return v;
}
static __device__ __forceinline__ float rmax16(float v) {
    v = fmaxf(v, __shfl_xor(v, 1, 32)); v = fmaxf(v, __shfl_xor(v, 2, 32));
    v = fmaxf(v, __shfl_xor(v, 4, 32)); v = fmaxf(v, __shfl_xor(v, 8, 32));
    return v;
}

// ---------------------------------------------------------------------------
// Pass 0: f32 -> bf16 conversion, 8 elements / thread-iteration (b128 I/O).
// ---------------------------------------------------------------------------
__global__ void k_cvt_bf16(const float* __restrict__ s, __bf16* __restrict__ d, int n) {
    int i = (blockIdx.x * blockDim.x + threadIdx.x) * 8;
    int stride = gridDim.x * blockDim.x * 8;
    for (; i < n; i += stride) {
        f32x4 a = *reinterpret_cast<const f32x4*>(s + i);
        f32x4 b = *reinterpret_cast<const f32x4*>(s + i + 4);
        u32x4 o;
        o.x = (uint32_t)bf16_bits(a.x) | ((uint32_t)bf16_bits(a.y) << 16);
        o.y = (uint32_t)bf16_bits(a.z) | ((uint32_t)bf16_bits(a.w) << 16);
        o.z = (uint32_t)bf16_bits(b.x) | ((uint32_t)bf16_bits(b.y) << 16);
        o.w = (uint32_t)bf16_bits(b.z) | ((uint32_t)bf16_bits(b.w) << 16);
        *reinterpret_cast<u32x4*>(d + i) = o;
    }
}

// ---------------------------------------------------------------------------
// Pass 1: fused QKV projection (+q_bias, +L2 norm, +temperature), bf16 out.
// grid = (B*L/64, H, 3), block = 32 (1 wave). z: 0=q 1=k 2=v(transposed).
// Each wave: 64 rows x 64 cols (one head), K-loop over 1024.
// ---------------------------------------------------------------------------
__global__ void __launch_bounds__(32, 1)
k_qkv(const __bf16* __restrict__ xb, const __bf16* __restrict__ yb,
      const __bf16* __restrict__ wqb, const __bf16* __restrict__ wkb,
      const __bf16* __restrict__ wvb,
      const float* __restrict__ q_bias, const float* __restrict__ scale_mul,
      __bf16* __restrict__ qo, __bf16* __restrict__ ko,
      __bf16* __restrict__ vto) {
    const int lane = threadIdx.x & 31;
    const int r16  = lane & 15;
    const int hi   = lane >> 4;
    const int m0   = blockIdx.x * 64;
    const int h    = blockIdx.y;
    const int z    = blockIdx.z;
    const int n0   = h * ND;

    const __bf16* inp = (z == 0) ? xb : yb;
    const __bf16* w   = (z == 0) ? wqb : ((z == 1) ? wkb : wvb);

    f32x8 acc[4][4] = {};
    for (int k0 = 0; k0 < NIN; k0 += 32) {
        bf16x16 a[4];
#pragma unroll
        for (int mi = 0; mi < 4; ++mi) {
            a[mi] = load_frag(inp + (size_t)(m0 + 16 * mi) * NIN + k0, NIN);
            // WGP-scope prefetch of the streamed A operand, 2 K-steps ahead
            __builtin_prefetch(inp + (size_t)(m0 + 16 * mi + r16) * NIN + k0 + 64 + hi * 8, 0, 3);
        }
#pragma unroll
        for (int j = 0; j < 4; ++j) {
            bf16x16 bfr = load_frag(w + (size_t)(n0 + 16 * j) * NIN + k0, NIN);
#pragma unroll
            for (int mi = 0; mi < 4; ++mi)
                acc[mi][j] = wmma_bf16(a[mi], bfr, acc[mi][j]);
        }
    }

    if (z == 0) {
#pragma unroll
        for (int j = 0; j < 4; ++j) {
            float qb = q_bias[n0 + 16 * j + r16];
#pragma unroll
            for (int mi = 0; mi < 4; ++mi)
#pragma unroll
                for (int r = 0; r < 8; ++r) acc[mi][j][r] += qb;
        }
    }

    if (z < 2) {
        float sc = (z == 0) ? __expf(fminf(scale_mul[h], MAXSM)) : 1.0f;
#pragma unroll
        for (int mi = 0; mi < 4; ++mi)
#pragma unroll
            for (int r = 0; r < 8; ++r) {
                float ss = 0.0f;
#pragma unroll
                for (int j = 0; j < 4; ++j) { float v = acc[mi][j][r]; ss += v * v; }
                ss = rsum16(ss);                   // full 64-wide row sum (per half)
                float inv = sc / fmaxf(sqrtf(ss), 1e-12f);
#pragma unroll
                for (int j = 0; j < 4; ++j) acc[mi][j][r] *= inv;
            }
    }

#pragma unroll
    for (int mi = 0; mi < 4; ++mi)
#pragma unroll
        for (int r = 0; r < 8; ++r) {
            const int gm = m0 + 16 * mi + r + 8 * hi;   // global row in [0, B*L)
            const int bb = gm >> 11;                    // batch
            const int l  = gm & (NL - 1);               // sequence pos
#pragma unroll
            for (int j = 0; j < 4; ++j) {
                __bf16 v = to_bf16(acc[mi][j][r]);
                const int d = 16 * j + r16;
                if (z < 2) {
                    __bf16* dst = (z == 0) ? qo : ko;   // [B,H,L,D]
                    dst[(((size_t)(bb * NH + h)) * NL + l) * ND + d] = v;
                } else {                                // [B,H,D,L] (transposed V)
                    vto[(((size_t)(bb * NH + h)) * ND + d) * NL + l] = v;
                }
            }
        }
}

// ---------------------------------------------------------------------------
// Pass 2: flash attention, 1 wave per 32-query tile, 32 keys per step.
// grid = (L/32, H, B), block = 32.
// ---------------------------------------------------------------------------
__global__ void __launch_bounds__(32, 1)
k_attn(const __bf16* __restrict__ qb, const __bf16* __restrict__ kb,
       const __bf16* __restrict__ vtb, const float* __restrict__ bias,
       __bf16* __restrict__ ob) {
    const int lane = threadIdx.x & 31;
    const int r16  = lane & 15;
    const int hi   = lane >> 4;
    const int q0   = blockIdx.x * 32;
    const int h    = blockIdx.y;
    const int bz   = blockIdx.z;

    const __bf16* Q  = qb  + ((size_t)(bz * NH + h)) * NL * ND;
    const __bf16* K  = kb  + ((size_t)(bz * NH + h)) * NL * ND;
    const __bf16* Vt = vtb + ((size_t)(bz * NH + h)) * ND * NL;

    __shared__ __bf16 lp[32 * 32];                 // P tiles staging (C->A layout)

    // Q fragments (2 x 16 queries x 64 dims), loaded once.
    bf16x16 qf[2][2];
#pragma unroll
    for (int mi = 0; mi < 2; ++mi) {
        qf[mi][0] = load_frag(Q + (size_t)(q0 + 16 * mi) * ND, ND);
        qf[mi][1] = load_frag(Q + (size_t)(q0 + 16 * mi) * ND + 32, ND);
    }

    f32x8 o[2][4] = {};
    float mrow[2][8], srow[2][8];
#pragma unroll
    for (int mi = 0; mi < 2; ++mi)
#pragma unroll
        for (int r = 0; r < 8; ++r) { mrow[mi][r] = -3.0e38f; srow[mi][r] = 0.0f; }

    for (int k0 = 0; k0 < NL; k0 += 32) {
        // ---- S = Q K^T for 32 keys (2 query tiles x 2 key tiles) ----
        f32x8 s[2][2] = {};
#pragma unroll
        for (int jj = 0; jj < 2; ++jj) {
            bf16x16 bk0 = load_frag(K + (size_t)(k0 + 16 * jj) * ND, ND);
            bf16x16 bk1 = load_frag(K + (size_t)(k0 + 16 * jj) * ND + 32, ND);
            // WGP-scope prefetch of the next key tile (streamed)
            __builtin_prefetch(K + (size_t)(k0 + 32 + 16 * jj + r16) * ND + hi * 8, 0, 3);
#pragma unroll
            for (int mi = 0; mi < 2; ++mi) {
                s[mi][jj] = wmma_bf16(qf[mi][0], bk0, s[mi][jj]);
                s[mi][jj] = wmma_bf16(qf[mi][1], bk1, s[mi][jj]);
            }
        }
        // ---- + attn_bias ----
#pragma unroll
        for (int mi = 0; mi < 2; ++mi)
#pragma unroll
            for (int jj = 0; jj < 2; ++jj)
#pragma unroll
                for (int r = 0; r < 8; ++r)
                    s[mi][jj][r] +=
                        bias[(size_t)(q0 + 16 * mi + r + 8 * hi) * NL + k0 + 16 * jj + r16];

        // ---- online softmax ----
#pragma unroll
        for (int mi = 0; mi < 2; ++mi)
#pragma unroll
            for (int r = 0; r < 8; ++r) {
                float vm = rmax16(fmaxf(s[mi][0][r], s[mi][1][r]));
                float mn = fmaxf(mrow[mi][r], vm);
                float al = __expf(mrow[mi][r] - mn);
                float p0 = __expf(s[mi][0][r] - mn);
                float p1 = __expf(s[mi][1][r] - mn);
                srow[mi][r] = srow[mi][r] * al + rsum16(p0 + p1);
                mrow[mi][r] = mn;
                s[mi][0][r] = p0; s[mi][1][r] = p1;
#pragma unroll
                for (int j = 0; j < 4; ++j) o[mi][j][r] *= al;
            }

        // ---- P: C layout -> A layout via LDS (wave-private, in-order DS) ----
#pragma unroll
        for (int mi = 0; mi < 2; ++mi)
#pragma unroll
            for (int jj = 0; jj < 2; ++jj)
#pragma unroll
                for (int r = 0; r < 8; ++r)
                    lp[(16 * mi + r + 8 * hi) * 32 + 16 * jj + r16] = to_bf16(s[mi][jj][r]);
        asm volatile("s_wait_dscnt 0" ::: "memory");

        bf16x16 pf[2];
#pragma unroll
        for (int mi = 0; mi < 2; ++mi) {
            union { bf16x16 v; u32x4 q[2]; } u;
            u.q[0] = *reinterpret_cast<const u32x4*>(&lp[(16 * mi + r16) * 32 + hi * 8]);
            u.q[1] = *reinterpret_cast<const u32x4*>(&lp[(16 * mi + r16) * 32 + hi * 8 + 16]);
            pf[mi] = u.v;
        }

        // ---- O += P V  (V^T rows are head-dims -> contiguous b128 loads) ----
#pragma unroll
        for (int j = 0; j < 4; ++j) {
            bf16x16 bv = load_frag(Vt + (size_t)(16 * j) * NL + k0, NL);
#pragma unroll
            for (int mi = 0; mi < 2; ++mi)
                o[mi][j] = wmma_bf16(pf[mi], bv, o[mi][j]);
        }
    }

    // ---- normalize + store O as bf16 [B, L, H*D] ----
#pragma unroll
    for (int mi = 0; mi < 2; ++mi)
#pragma unroll
        for (int r = 0; r < 8; ++r) {
            const int gq = q0 + 16 * mi + r + 8 * hi;
            float inv = 1.0f / srow[mi][r];
#pragma unroll
            for (int j = 0; j < 4; ++j) {
                ob[((size_t)bz * NL + gq) * NIN + h * ND + 16 * j + r16] =
                    to_bf16(o[mi][j][r] * inv);
            }
        }
}

// ---------------------------------------------------------------------------
// Pass 3: output projection, f32 out + p_bias.
// grid = (B*L/64, 16), block 32; 64x64 tile per wave.
// ---------------------------------------------------------------------------
__global__ void __launch_bounds__(32, 1)
k_oproj(const __bf16* __restrict__ ob, const __bf16* __restrict__ wpb,
        const float* __restrict__ p_bias, float* __restrict__ out) {
    const int lane = threadIdx.x & 31;
    const int r16  = lane & 15;
    const int hi   = lane >> 4;
    const int m0   = blockIdx.x * 64;
    const int n0   = blockIdx.y * ND;

    f32x8 acc[4][4] = {};
    for (int k0 = 0; k0 < NIN; k0 += 32) {
        bf16x16 a[4];
#pragma unroll
        for (int mi = 0; mi < 4; ++mi) {
            a[mi] = load_frag(ob + (size_t)(m0 + 16 * mi) * NIN + k0, NIN);
            __builtin_prefetch(ob + (size_t)(m0 + 16 * mi + r16) * NIN + k0 + 64 + hi * 8, 0, 3);
        }
#pragma unroll
        for (int j = 0; j < 4; ++j) {
            bf16x16 bfr = load_frag(wpb + (size_t)(n0 + 16 * j) * NIN + k0, NIN);
#pragma unroll
            for (int mi = 0; mi < 4; ++mi)
                acc[mi][j] = wmma_bf16(a[mi], bfr, acc[mi][j]);
        }
    }
#pragma unroll
    for (int mi = 0; mi < 4; ++mi)
#pragma unroll
        for (int r = 0; r < 8; ++r) {
            const int gm = m0 + 16 * mi + r + 8 * hi;
#pragma unroll
            for (int j = 0; j < 4; ++j) {
                const int n = n0 + 16 * j + r16;
                out[(size_t)gm * NIN + n] = acc[mi][j][r] + p_bias[n];
            }
        }
}

// ---------------------------------------------------------------------------
// Host launcher
// ---------------------------------------------------------------------------
extern "C" void kernel_launch(void* const* d_in, const int* in_sizes, int n_in,
                              void* d_out, int out_size, void* d_ws, size_t ws_size,
                              hipStream_t stream) {
    (void)in_sizes; (void)n_in; (void)out_size; (void)ws_size;

    const float* x         = (const float*)d_in[0];
    const float* y         = (const float*)d_in[1];
    const float* attn_bias = (const float*)d_in[2];
    const float* Wq        = (const float*)d_in[3];
    const float* Wk        = (const float*)d_in[4];
    const float* Wv        = (const float*)d_in[5];
    const float* q_bias    = (const float*)d_in[6];
    const float* scale_mul = (const float*)d_in[7];
    const float* Wp        = (const float*)d_in[8];
    const float* p_bias    = (const float*)d_in[9];
    float*       out       = (float*)d_out;

    const size_t NTOK = (size_t)NB * NL;           // 8192
    const size_t SZ_X = NTOK * NIN;                // 8.4M elems
    const size_t SZ_W = (size_t)NIN * NIN;         // 1M elems

    char* ws = (char*)d_ws;
    size_t off = 0;
    __bf16* xb  = (__bf16*)(ws + off); off += SZ_X * 2;
    __bf16* yb  = (__bf16*)(ws + off); off += SZ_X * 2;
    __bf16* wqb = (__bf16*)(ws + off); off += SZ_W * 2;
    __bf16* wkb = (__bf16*)(ws + off); off += SZ_W * 2;
    __bf16* wvb = (__bf16*)(ws + off); off += SZ_W * 2;
    __bf16* wpb = (__bf16*)(ws + off); off += SZ_W * 2;
    __bf16* qbf = (__bf16*)(ws + off); off += SZ_X * 2;   // [B,H,L,D]
    __bf16* kbf = (__bf16*)(ws + off); off += SZ_X * 2;   // [B,H,L,D]
    __bf16* vtb = (__bf16*)(ws + off); off += SZ_X * 2;   // [B,H,D,L]
    __bf16* obf = (__bf16*)(ws + off); off += SZ_X * 2;   // [B,L,H*D]

    // Pass 0: conversions (8 elems/thread/iter)
    k_cvt_bf16<<<2048, 256, 0, stream>>>(x,  xb,  (int)SZ_X);
    k_cvt_bf16<<<2048, 256, 0, stream>>>(y,  yb,  (int)SZ_X);
    k_cvt_bf16<<<512, 256, 0, stream>>>(Wq, wqb, (int)SZ_W);
    k_cvt_bf16<<<512, 256, 0, stream>>>(Wk, wkb, (int)SZ_W);
    k_cvt_bf16<<<512, 256, 0, stream>>>(Wv, wvb, (int)SZ_W);
    k_cvt_bf16<<<512, 256, 0, stream>>>(Wp, wpb, (int)SZ_W);

    // Pass 1: QKV projections (+norm/temperature)
    k_qkv<<<dim3(NTOK / 64, NH, 3), 32, 0, stream>>>(
        xb, yb, wqb, wkb, wvb, q_bias, scale_mul, qbf, kbf, vtb);

    // Pass 2: attention
    k_attn<<<dim3(NL / 32, NH, NB), 32, 0, stream>>>(qbf, kbf, vtb, attn_bias, obf);

    // Pass 3: output projection
    k_oproj<<<dim3(NTOK / 64, NIN / ND), 32, 0, stream>>>(obf, wpb, p_bias, out);
}
